// MultiHeadAttention_72456098284091
// MI455X (gfx1250) — matrix operand order repaired and checked by reading, not compiled, not run on hardware
//
#include <hip/hip_runtime.h>
#include <hip/hip_bf16.h>

// MHA forward for MI455X (gfx1250), bf16 WMMA with f32 accumulation.
// Pipeline: [qkv_gemm] -> [flash attention] -> [out_gemm], all on `stream`.
// GEMMs use ping-pong LDS double buffering + register prefetch so global
// loads for chunk i+1 overlap the 4 back-to-back WMMAs of chunk i.

typedef __attribute__((ext_vector_type(16))) __bf16 v16bf;
typedef __attribute__((ext_vector_type(8)))  __bf16 v8bf;
typedef __attribute__((ext_vector_type(8)))  float  v8f;
typedef __attribute__((ext_vector_type(4)))  float  v4f;

static constexpr int HID  = 1024;
static constexpr int SEQ  = 2048;
static constexpr int NB   = 2;     // batch
static constexpr int NH   = 16;    // heads
static constexpr int HD   = 64;    // head dim
static constexpr int PIT  = 40;    // LDS row pitch in bf16 (80B, 16B aligned)

union FragU { v16bf v; v8bf h[2]; };

// A-matrix 16x32 bf16 fragment (this file's k order, not ISA table order): lane = {row = lane&15, half = lane>>4},
// holds K = 16*half .. 16*half+15 (contiguous).
__device__ __forceinline__ v16bf ld_frag_a(const __bf16* base, int lane, int pitch) {
    const int row = lane & 15, half = lane >> 4;
    const __bf16* p = base + row * pitch + half * 16;
    FragU u;
    u.h[0] = *(const v8bf*)(p);
    u.h[1] = *(const v8bf*)(p + 8);
    return u.v;
}

// B-matrix 32x16 bf16 fragment: lane = {col = lane&15, half = lane>>4},
// holds K = 16*half .. 16*half+15 (contiguous). `base` is tile stored [col][k].
__device__ __forceinline__ v16bf ld_frag_b(const __bf16* base, int lane, int pitch) {
    const int col = lane & 15, half = lane >> 4;
    const __bf16* p = base + col * pitch + half * 16;
    FragU u;
    u.h[0] = *(const v8bf*)(p);
    u.h[1] = *(const v8bf*)(p + 8);
    return u.v;
}

// One 16(M)x64(N)x32(K) step: 1 A fragment, 4 B fragments loaded into
// distinct registers (pipelined ds_loads), then 4 back-to-back WMMAs.
__device__ __forceinline__ void compute_step(const __bf16* lA, const __bf16* lB,
                                             int wave, int lane, v8f acc[4]) {
    v16bf a  = ld_frag_a(lA + wave * 16 * PIT, lane, PIT);
    v16bf b0 = ld_frag_b(lB + 0 * 16 * PIT, lane, PIT);
    v16bf b1 = ld_frag_b(lB + 1 * 16 * PIT, lane, PIT);
    v16bf b2 = ld_frag_b(lB + 2 * 16 * PIT, lane, PIT);
    v16bf b3 = ld_frag_b(lB + 3 * 16 * PIT, lane, PIT);
    acc[0] = __builtin_amdgcn_wmma_f32_16x16x32_bf16(false, a, false, b0, (short)0, acc[0], false, false);
    acc[1] = __builtin_amdgcn_wmma_f32_16x16x32_bf16(false, a, false, b1, (short)0, acc[1], false, false);
    acc[2] = __builtin_amdgcn_wmma_f32_16x16x32_bf16(false, a, false, b2, (short)0, acc[2], false, false);
    acc[3] = __builtin_amdgcn_wmma_f32_16x16x32_bf16(false, a, false, b3, (short)0, acc[3], false, false);
}

__device__ __forceinline__ void cvt_store16(__bf16* d, const v4f* t4) {
#pragma unroll
    for (int i = 0; i < 4; ++i) {
        d[4 * i + 0] = (__bf16)t4[i][0];
        d[4 * i + 1] = (__bf16)t4[i][1];
        d[4 * i + 2] = (__bf16)t4[i][2];
        d[4 * i + 3] = (__bf16)t4[i][3];
    }
}

// ---------------------------------------------------------------------------
// Kernel 1: fused QKV projection.  y = x @ W^T + b, f32 inputs -> bf16 outputs.
// Block = 128 threads (4 waves), tile 64(M) x 64(N), K-step 32, ping-pong LDS.
// grid.x: 48 column tiles (16 per matrix: Q,K,V), grid.y: 64 row tiles.
// Q,K written [b,h,n,d]; V written transposed [b,h,d,n] for the PV matmul.
// ---------------------------------------------------------------------------
__global__ __launch_bounds__(128) void qkv_gemm_kernel(
    const float* __restrict__ x,
    const float* __restrict__ Wq, const float* __restrict__ bq,
    const float* __restrict__ Wk, const float* __restrict__ bk,
    const float* __restrict__ Wv, const float* __restrict__ bv,
    __bf16* __restrict__ qbuf, __bf16* __restrict__ kbuf, __bf16* __restrict__ vbuf)
{
    __shared__ __align__(16) __bf16 lA[2][64 * PIT];
    __shared__ __align__(16) __bf16 lB[2][64 * PIT];

    const int tid  = threadIdx.x;
    const int lane = tid & 31;
    const int wave = tid >> 5;
    const int mBase = blockIdx.y * 64;
    const int which = blockIdx.x >> 4;            // 0=Q 1=K 2=V
    const int nBase = (blockIdx.x & 15) * 64;     // within [0,1024)

    const float* W    = (which == 0) ? Wq : (which == 1) ? Wk : Wv;
    const float* bias = (which == 0) ? bq : (which == 1) ? bk : bv;

    v8f acc[4] = {};

    const int r  = tid >> 1;            // 0..63: tile row handled by this thread
    const int cs = (tid & 1) * 16;      // 0 or 16: k sub-segment
    const float* srcA = x + (size_t)(mBase + r) * HID + cs;
    const float* srcB = W + (size_t)(nBase + r) * HID + cs;

    v4f pa[4], pb[4];                   // register prefetch staging
#pragma unroll
    for (int i = 0; i < 4; ++i) {       // chunk 0
        pa[i] = *(const v4f*)(srcA + 4 * i);
        pb[i] = *(const v4f*)(srcB + 4 * i);
    }
    cvt_store16(lA[0] + r * PIT + cs, pa);
    cvt_store16(lB[0] + r * PIT + cs, pb);
    __syncthreads();

    for (int it = 0; it < HID / 32; ++it) {
        if (it + 1 < HID / 32) {        // prefetch chunk it+1 into registers
            const int kc = (it + 1) * 32;
#pragma unroll
            for (int i = 0; i < 4; ++i) {
                pa[i] = *(const v4f*)(srcA + kc + 4 * i);
                pb[i] = *(const v4f*)(srcB + kc + 4 * i);
            }
        }
        compute_step(lA[it & 1], lB[it & 1], wave, lane, acc);
        if (it + 1 < HID / 32) {        // fill the other buffer, one barrier/iter
            cvt_store16(lA[(it + 1) & 1] + r * PIT + cs, pa);
            cvt_store16(lB[(it + 1) & 1] + r * PIT + cs, pb);
            __syncthreads();
        }
    }

    // Epilogue: bias add, f32 -> bf16, scatter to head-major layouts.
    const int half = lane >> 4, c = lane & 15;
#pragma unroll
    for (int t = 0; t < 4; ++t) {
        const int nloc = nBase + t * 16 + c;      // column within this matrix
        const float bval = bias[nloc];
        const int h = nloc >> 6, d = nloc & 63;
#pragma unroll
        for (int rr = 0; rr < 8; ++rr) {
            const int token = mBase + wave * 16 + rr + half * 8;
            const int b  = token >> 11;           // /2048
            const int tn = token & 2047;
            const float val = acc[t][rr] + bval;
            if (which == 0)
                qbuf[(((size_t)(b * NH + h)) * SEQ + tn) * HD + d] = (__bf16)val;
            else if (which == 1)
                kbuf[(((size_t)(b * NH + h)) * SEQ + tn) * HD + d] = (__bf16)val;
            else
                vbuf[(((size_t)(b * NH + h)) * HD + d) * SEQ + tn] = (__bf16)val;
        }
    }
}

// ---------------------------------------------------------------------------
// Kernel 2: flash-style attention. One wave per 16-query tile.
// Block = 256 threads (8 waves); grid = B*H*(SEQ/16)/8 = 512 blocks.
// S = Q K^T via 2 x wmma(K=32); online softmax; O += P V via 4 x wmma per
// 32-key chunk (P re-laid out C->A through a per-wave LDS scratch).
// global_prefetch of the next 32-key K/V chunk overlaps softmax VALU work.
// ---------------------------------------------------------------------------
__global__ __launch_bounds__(256) void attn_kernel(
    const __bf16* __restrict__ q,   // [b,h,n,d]
    const __bf16* __restrict__ k,   // [b,h,n,d]
    const __bf16* __restrict__ vT,  // [b,h,d,n]
    __bf16* __restrict__ ao)        // [b,n, h*64+d]  (token-major for out proj)
{
    __shared__ __align__(16) __bf16 pst_all[8][16 * PIT];

    const int lane = threadIdx.x & 31;
    const int wave = threadIdx.x >> 5;
    __bf16* pst = pst_all[wave];

    const int tile = blockIdx.x * 8 + wave;   // 0..4095
    const int qt = tile & 127;                // query tile within head
    const int bh = tile >> 7;                 // b*16 + h
    const int half = lane >> 4, c16 = lane & 15;

    // Q A-fragments for D=64 split into two K=32 fragments.
    const __bf16* qp = q + ((size_t)bh * SEQ + qt * 16 + c16) * HD;
    v16bf aq[2];
#pragma unroll
    for (int f = 0; f < 2; ++f) {
        const __bf16* p = qp + f * 32 + half * 16;
        FragU u;
        u.h[0] = *(const v8bf*)(p);
        u.h[1] = *(const v8bf*)(p + 8);
        aq[f] = u.v;
    }

    const __bf16* kbase = k + (size_t)bh * SEQ * HD;
    const __bf16* vbase = vT + ((size_t)bh * HD + c16) * SEQ;

    float m_[8], l_[8];
#pragma unroll
    for (int rr = 0; rr < 8; ++rr) { m_[rr] = -3.0e38f; l_[rr] = 0.0f; }
    v8f o[4] = {};

    for (int j = 0; j < SEQ / 32; ++j) {
        if (j + 1 < SEQ / 32) {
            // Prefetch next chunk's K rows and V columns (global_prefetch_b8).
            __builtin_prefetch(kbase + (size_t)((j + 1) * 32 + c16) * HD + half * 32, 0, 3);
            __builtin_prefetch(vbase + (j + 1) * 32 + half * 16, 0, 3);
        }
#pragma unroll
        for (int js = 0; js < 2; ++js) {
            const int key0 = j * 32 + js * 16;
            // K^T B-fragments: B[d,key] = K[key,d]; lane col = key0+c16,
            // per-fragment K-rows d = f*32 + 16*half .. +15 (contiguous in k).
            const __bf16* kp = kbase + (size_t)(key0 + c16) * HD;
            FragU kb0, kb1;
            kb0.h[0] = *(const v8bf*)(kp + half * 16);
            kb0.h[1] = *(const v8bf*)(kp + half * 16 + 8);
            kb1.h[0] = *(const v8bf*)(kp + 32 + half * 16);
            kb1.h[1] = *(const v8bf*)(kp + 32 + half * 16 + 8);
            v8f s8 = {};
            s8 = __builtin_amdgcn_wmma_f32_16x16x32_bf16(
                false, aq[0], false, kb0.v, (short)0, s8, false, false);
            s8 = __builtin_amdgcn_wmma_f32_16x16x32_bf16(
                false, aq[1], false, kb1.v, (short)0, s8, false, false);
            // Online softmax. Row M = rr + 8*half lives on the 16 lanes of
            // this half; xor masks 1..8 reduce within the half.
#pragma unroll
            for (int rr = 0; rr < 8; ++rr) {
                float s = s8[rr] * 0.125f;   // 1/sqrt(64)
                float mx = s;
#pragma unroll
                for (int mm = 1; mm <= 8; mm <<= 1)
                    mx = fmaxf(mx, __shfl_xor(mx, mm, 32));
                const float mnew  = fmaxf(m_[rr], mx);
                const float alpha = __expf(m_[rr] - mnew);
                const float p     = __expf(s - mnew);
                float rsum = p;
#pragma unroll
                for (int mm = 1; mm <= 8; mm <<= 1)
                    rsum += __shfl_xor(rsum, mm, 32);
                l_[rr] = l_[rr] * alpha + rsum;
                m_[rr] = mnew;
#pragma unroll
                for (int t = 0; t < 4; ++t) o[t][rr] *= alpha;
                // Stage P into A-layout scratch (C-layout -> LDS -> A-frag).
                pst[(rr + half * 8) * PIT + js * 16 + c16] = (__bf16)p;
            }
        }
        // P A-fragment (16 queries x 32 keys), same-wave LDS RAW (DScnt).
        FragU up;
        {
            const __bf16* p = pst + c16 * PIT + half * 16;
            up.h[0] = *(const v8bf*)(p);
            up.h[1] = *(const v8bf*)(p + 8);
        }
        // V B-fragments: 4 distinct register sets loaded first, then 4 WMMAs.
        FragU uv[4];
#pragma unroll
        for (int t = 0; t < 4; ++t) {
            const __bf16* vp = vbase + (size_t)(t * 16) * SEQ + j * 32 + half * 16;
            uv[t].h[0] = *(const v8bf*)(vp);
            uv[t].h[1] = *(const v8bf*)(vp + 8);
        }
#pragma unroll
        for (int t = 0; t < 4; ++t) {
            o[t] = __builtin_amdgcn_wmma_f32_16x16x32_bf16(
                false, up.v, false, uv[t].v, (short)0, o[t], false, false);
        }
    }

    // Normalize and scatter to token-major bf16 for the output projection.
    const int b = bh >> 4, h = bh & 15;
#pragma unroll
    for (int rr = 0; rr < 8; ++rr) {
        const float inv = 1.0f / l_[rr];
        const int token = qt * 16 + rr + half * 8;
#pragma unroll
        for (int t = 0; t < 4; ++t) {
            ao[((size_t)(b * SEQ + token)) * HID + h * HD + t * 16 + c16] =
                (__bf16)(o[t][rr] * inv);
        }
    }
}

// ---------------------------------------------------------------------------
// Kernel 3: output projection. out = attn_out(bf16) @ Wo^T + bo -> f32.
// Same ping-pong tiling as kernel 1; A tile is already bf16.
// grid.x: 16 column tiles, grid.y: 64 row tiles.
// ---------------------------------------------------------------------------
__global__ __launch_bounds__(128) void out_gemm_kernel(
    const __bf16* __restrict__ ao, const float* __restrict__ Wo,
    const float* __restrict__ bo, float* __restrict__ out)
{
    __shared__ __align__(16) __bf16 lA[2][64 * PIT];
    __shared__ __align__(16) __bf16 lB[2][64 * PIT];

    const int tid  = threadIdx.x;
    const int lane = tid & 31;
    const int wave = tid >> 5;
    const int mBase = blockIdx.y * 64;
    const int nBase = blockIdx.x * 64;

    v8f acc[4] = {};
    const int r  = tid >> 1;
    const int cs = (tid & 1) * 16;
    const __bf16* srcA = ao + (size_t)(mBase + r) * HID + cs;
    const float*  srcB = Wo + (size_t)(nBase + r) * HID + cs;

    v8bf par[2];
    v4f  pb[4];
    par[0] = *(const v8bf*)(srcA);
    par[1] = *(const v8bf*)(srcA + 8);
#pragma unroll
    for (int i = 0; i < 4; ++i) pb[i] = *(const v4f*)(srcB + 4 * i);
    {
        __bf16* dA = lA[0] + r * PIT + cs;
        *(v8bf*)(dA) = par[0]; *(v8bf*)(dA + 8) = par[1];
        cvt_store16(lB[0] + r * PIT + cs, pb);
    }
    __syncthreads();

    for (int it = 0; it < HID / 32; ++it) {
        if (it + 1 < HID / 32) {
            const int kc = (it + 1) * 32;
            par[0] = *(const v8bf*)(srcA + kc);
            par[1] = *(const v8bf*)(srcA + kc + 8);
#pragma unroll
            for (int i = 0; i < 4; ++i) pb[i] = *(const v4f*)(srcB + kc + 4 * i);
        }
        compute_step(lA[it & 1], lB[it & 1], wave, lane, acc);
        if (it + 1 < HID / 32) {
            __bf16* dA = lA[(it + 1) & 1] + r * PIT + cs;
            *(v8bf*)(dA) = par[0]; *(v8bf*)(dA + 8) = par[1];
            cvt_store16(lB[(it + 1) & 1] + r * PIT + cs, pb);
            __syncthreads();
        }
    }

    const int half = lane >> 4, c = lane & 15;
#pragma unroll
    for (int t = 0; t < 4; ++t) {
        const int n = nBase + t * 16 + c;
        const float bval = bo[n];
#pragma unroll
        for (int rr = 0; rr < 8; ++rr) {
            const int token = mBase + wave * 16 + rr + half * 8;
            out[(size_t)token * HID + n] = acc[t][rr] + bval;
        }
    }
}

// ---------------------------------------------------------------------------
extern "C" void kernel_launch(void* const* d_in, const int* in_sizes, int n_in,
                              void* d_out, int out_size, void* d_ws, size_t ws_size,
                              hipStream_t stream)
{
    const float* x  = (const float*)d_in[0];
    const float* Wq = (const float*)d_in[1];
    const float* bq = (const float*)d_in[2];
    const float* Wk = (const float*)d_in[3];
    const float* bk = (const float*)d_in[4];
    const float* Wv = (const float*)d_in[5];
    const float* bv = (const float*)d_in[6];
    const float* Wo = (const float*)d_in[7];
    const float* bo = (const float*)d_in[8];
    float* out = (float*)d_out;

    // Workspace: Q, K (head-major), V (transposed), attn_out — each 8 MiB bf16.
    char* ws = (char*)d_ws;
    const size_t BUF = (size_t)NB * NH * SEQ * HD * sizeof(__bf16);  // 8 MiB
    __bf16* qb  = (__bf16*)(ws);
    __bf16* kb  = (__bf16*)(ws + BUF);
    __bf16* vb  = (__bf16*)(ws + 2 * BUF);
    __bf16* aob = (__bf16*)(ws + 3 * BUF);

    qkv_gemm_kernel<<<dim3(48, 64), 128, 0, stream>>>(
        x, Wq, bq, Wk, bk, Wv, bv, qb, kb, vb);
    attn_kernel<<<dim3(512), 256, 0, stream>>>(qb, kb, vb, aob);
    out_gemm_kernel<<<dim3(16, 64), 128, 0, stream>>>(aob, Wo, bo, out);
}
